// RNN_33122787787246
// MI455X (gfx1250) — matrix operand order repaired and checked by reading, not compiled, not run on hardware
//
#include <hip/hip_runtime.h>
#include <hip/hip_bf16.h>
#include <math.h>

// Problem constants (match reference): T=512, B=64, I=512, L=1024
#define T_STEPS 512
#define BATCH   64
#define I_DIM   512
#define L_DIM   1024

typedef __bf16 bf16;
typedef __bf16 bf16x16 __attribute__((ext_vector_type(16)));
typedef __bf16 bf16x8  __attribute__((ext_vector_type(8)));
typedef float  f32x8   __attribute__((ext_vector_type(8)));
typedef float  f32x4   __attribute__((ext_vector_type(4)));

// gfx1250 async global->LDS copy path (ASYNCcnt), if the toolchain exposes it.
#if defined(__HIP_DEVICE_COMPILE__) &&                                         \
    __has_builtin(__builtin_amdgcn_global_load_async_to_lds_b128) &&           \
    __has_builtin(__builtin_amdgcn_s_wait_asynccnt)
#define HAVE_ASYNC_COPY 1
// Exact parameter pointee type per the compiler diagnostic:
//   int __attribute__((vector_size(4*sizeof(int))))
typedef int async_v4i __attribute__((vector_size(4 * sizeof(int))));
#define ASYNC_GSRC(p) ((__attribute__((address_space(1))) async_v4i*)(p))
#define ASYNC_LDST(p) ((__attribute__((address_space(3))) async_v4i*)(p))
#else
#define HAVE_ASYNC_COPY 0
#endif

// ---------------------------------------------------------------------------
// Fragment loaders for V_WMMA_F32_16X16X32_BF16 (wave32).
//
// A (16x32, MxK), contiguous k order used by this code (B matches):
//   lane = m + 16*g ; element e=0..7  -> K = 16*g + e
//                     element e=8..15 -> K = 16*g + e
// Both runs are 8 contiguous fp32 of the source row -> f32x4 vector loads,
// converted inline with v_cvt_pk_bf16_f32.
// ---------------------------------------------------------------------------
__device__ __forceinline__ bf16x16 load_a_from_f32(const float* __restrict__ row,
                                                   int kc, int g) {
    const f32x4* p0 = (const f32x4*)(row + kc + 16 * g);       // K = kc+16g+0..7
    const f32x4* p1 = (const f32x4*)(row + kc + 16 * g + 8);   // K = kc+16g+8..15
    f32x4 a0 = p0[0], a1 = p0[1];
    f32x4 a2 = p1[0], a3 = p1[1];
    bf16x16 a;
#pragma unroll
    for (int j = 0; j < 4; ++j) {
        a[j]      = (bf16)a0[j];
        a[4 + j]  = (bf16)a1[j];
        a[8 + j]  = (bf16)a2[j];
        a[12 + j] = (bf16)a3[j];
    }
    return a;
}

// ---------------------------------------------------------------------------
// B (32x16, KxN) with B[k][n] = W[n][k] (W row-major [N,K]):
//   lane = n + 16*g ; element e -> K = 16*g + e
// 16 contiguous bf16 from row n -> two 16-byte vector loads (global or LDS).
// ---------------------------------------------------------------------------
__device__ __forceinline__ bf16x16 load_b_from_w(const bf16* __restrict__ w,
                                                 int n, int ld, int kc, int g) {
    const bf16x8* p = (const bf16x8*)(w + (size_t)n * ld + kc + 16 * g);
    bf16x8 lo = p[0];
    bf16x8 hi = p[1];
    bf16x16 b;
#pragma unroll
    for (int j = 0; j < 8; ++j) {
        b[j]     = lo[j];
        b[8 + j] = hi[j];
    }
    return b;
}

// Same fragment pull, but from an LDS tile laid out [64 rows][32 bf16].
__device__ __forceinline__ bf16x16 load_b_from_lds(const bf16* tile, int row, int g) {
    const bf16x8* p = (const bf16x8*)(tile + row * 32 + 16 * g);
    bf16x8 lo = p[0];
    bf16x8 hi = p[1];
    bf16x16 b;
#pragma unroll
    for (int j = 0; j < 8; ++j) {
        b[j]     = lo[j];
        b[8 + j] = hi[j];
    }
    return b;
}

// ---------------------------------------------------------------------------
// fp32 -> bf16 weight conversion (run once per call; Wi and Wh).
// ---------------------------------------------------------------------------
__global__ void f32_to_bf16_kernel(const float* __restrict__ in,
                                   bf16* __restrict__ out, int n) {
    int i = blockIdx.x * 256 + threadIdx.x;
    if (i < n) out[i] = (bf16)in[i];
}

// ---------------------------------------------------------------------------
// Phase 1: xi[m,n] = sum_k x[m,k]*Wi[n,k] + bi[n], written into d_out.
// Block = 128 threads (4 waves); block tile = 64(M) x 64(N); each wave owns
// a 16(M) row strip. The Wi B-tile (64 rows x 32 K = 4 KB) is shared by all
// 4 waves, so it is staged through a double-buffered LDS tile using async
// global->LDS copies (overlapped with the current chunk's WMMAs).
// grid = (M/64, N/64) = (512, 16)
// ---------------------------------------------------------------------------
__global__ __launch_bounds__(128) void rnn_xproj_kernel(
    const float* __restrict__ x,     // [M, I_DIM] fp32
    const bf16*  __restrict__ wib,   // [L_DIM, I_DIM] bf16
    const float* __restrict__ bi,    // [L_DIM]
    float* __restrict__ xi_out) {    // [M, L_DIM] (= d_out)
    __shared__ __align__(16) bf16 btile[2][64 * 32];   // 2 x 4 KB

    const int tid  = threadIdx.x;
    const int lane = tid & 31;
    const int wave = tid >> 5;
    const int g    = lane >> 4;
    const int ln   = lane & 15;
    const int m0   = (blockIdx.x * 4 + wave) * 16;
    const int nb   = blockIdx.y * 64;

    // Cooperative B-tile staging: thread copies 32 bytes (16 bf16).
    const int crow  = tid >> 1;                    // 0..63 (row of tile)
    const int chalf = (tid & 1) * 16;              // 0 or 16 (k offset)
    const bf16* gsrc = wib + (size_t)(nb + crow) * I_DIM + chalf;
    bf16* ldst[2] = { &btile[0][crow * 32 + chalf], &btile[1][crow * 32 + chalf] };

    f32x8 acc[4];
#pragma unroll
    for (int tt = 0; tt < 4; ++tt)
#pragma unroll
        for (int e = 0; e < 8; ++e) acc[tt][e] = 0.0f;

    const float* arow = x + (size_t)(m0 + ln) * I_DIM;

    // Stage chunk 0, load A fragment 0.
#if HAVE_ASYNC_COPY
    __builtin_amdgcn_global_load_async_to_lds_b128(
        ASYNC_GSRC(gsrc), ASYNC_LDST(ldst[0]), 0, 0);
    __builtin_amdgcn_global_load_async_to_lds_b128(
        ASYNC_GSRC(gsrc), ASYNC_LDST(ldst[0]), 16, 0);
    __builtin_amdgcn_s_wait_asynccnt(0);
#else
    {
        const bf16x8* s = (const bf16x8*)gsrc;
        bf16x8* d = (bf16x8*)ldst[0];
        d[0] = s[0];
        d[1] = s[1];
    }
#endif
    bf16x16 a_cur = load_a_from_f32(arow, 0, g);
    __syncthreads();

    const int NCHUNK = I_DIM / 32;  // 16
    for (int i = 0; i < NCHUNK; ++i) {
        const int buf = i & 1;
        bf16x16 a_nxt;
        if (i + 1 < NCHUNK) {
            const int kn = (i + 1) * 32;
            // Kick off next tile's async copy; completion waited at loop end.
#if HAVE_ASYNC_COPY
            __builtin_amdgcn_global_load_async_to_lds_b128(
                ASYNC_GSRC(gsrc + kn), ASYNC_LDST(ldst[buf ^ 1]), 0, 0);
            __builtin_amdgcn_global_load_async_to_lds_b128(
                ASYNC_GSRC(gsrc + kn), ASYNC_LDST(ldst[buf ^ 1]), 16, 0);
#else
            const bf16x8* s = (const bf16x8*)(gsrc + kn);
            bf16x8* d = (bf16x8*)ldst[buf ^ 1];
            d[0] = s[0];
            d[1] = s[1];
#endif
            a_nxt = load_a_from_f32(arow, kn, g);
        }
#pragma unroll
        for (int tt = 0; tt < 4; ++tt) {
            bf16x16 b = load_b_from_lds(&btile[buf][0], tt * 16 + ln, g);
            acc[tt] = __builtin_amdgcn_wmma_f32_16x16x32_bf16(
                false, a_cur, false, b, (short)0, acc[tt], false, false);
        }
        a_cur = a_nxt;
#if HAVE_ASYNC_COPY
        __builtin_amdgcn_s_wait_asynccnt(0);
#endif
        __syncthreads();
    }

#pragma unroll
    for (int tt = 0; tt < 4; ++tt) {
        const int n = nb + tt * 16 + ln;
        const float bias = bi[n];
#pragma unroll
        for (int r = 0; r < 8; ++r) {
            const int m = m0 + r + 8 * g;  // C/D layout: vgpr r -> row r + 8g
            xi_out[(size_t)m * L_DIM + n] = acc[tt][r] + bias;
        }
    }
}

// ---------------------------------------------------------------------------
// Phase 2 (one launch per timestep t):
//   d_out[t] = tanh(d_out[t] + h_prev @ Wh^T),  h_prev = d_out[t-1] (or h0).
// Block = 256 threads (8 waves). Block tile = 16(M) x 64(N).
// K=1024 split 8 ways across the waves (shorter serial WMMA chain for this
// latency-bound GEMM); A/B fragments are register double-buffered so the
// next chunk's loads overlap the current chunk's WMMAs. Partials are
// reduced through LDS, then each wave finalizes (tanh + in-place update)
// a (tile, row-group) slice. grid = (B/16, L/64) = (4, 16).
// ---------------------------------------------------------------------------
__global__ __launch_bounds__(256) void rnn_step_kernel(
    const bf16*  __restrict__ whb,   // [L_DIM, L_DIM] bf16
    const float* __restrict__ h0,    // [BATCH, L_DIM] initial hidden state
    float* __restrict__ out,         // [T_STEPS, BATCH, L_DIM] (= d_out)
    int t) {
    __shared__ float red[8 * 1024];  // 8 waves x (4 tiles x 8 rows x 32 lanes)

    const int tid  = threadIdx.x;
    const int lane = tid & 31;
    const int wave = tid >> 5;
    const int g    = lane >> 4;
    const int ln   = lane & 15;
    const int m0   = blockIdx.x * 16;
    const int nb   = blockIdx.y * 64;

    float* xo = out + (size_t)t * BATCH * L_DIM;

    // Warm the xi tile this block will read-modify-write in the epilogue.
    __builtin_prefetch(xo + (size_t)(m0 + (tid >> 4)) * L_DIM + nb + (tid & 15) * 4, 0, 3);

    const float* hbase = (t == 0) ? h0 : out + (size_t)(t - 1) * BATCH * L_DIM;
    const float* hrow  = hbase + (size_t)(m0 + ln) * L_DIM;

    f32x8 acc[4];
#pragma unroll
    for (int tt = 0; tt < 4; ++tt)
#pragma unroll
        for (int e = 0; e < 8; ++e) acc[tt][e] = 0.0f;

    const int kbeg = wave * (L_DIM / 8);           // 128-wide K slice per wave

    // Software pipeline: preload chunk 0, then prefetch i+1 while doing WMMA i.
    bf16x16 a_cur = load_a_from_f32(hrow, kbeg, g);
    bf16x16 b_cur[4];
#pragma unroll
    for (int tt = 0; tt < 4; ++tt)
        b_cur[tt] = load_b_from_w(whb, nb + tt * 16 + ln, L_DIM, kbeg, g);

#pragma unroll
    for (int ic = 0; ic < 4; ++ic) {               // 4 chunks of K=32
        bf16x16 a_nxt;
        bf16x16 b_nxt[4];
        if (ic < 3) {
            const int kn = kbeg + (ic + 1) * 32;
            a_nxt = load_a_from_f32(hrow, kn, g);
#pragma unroll
            for (int tt = 0; tt < 4; ++tt)
                b_nxt[tt] = load_b_from_w(whb, nb + tt * 16 + ln, L_DIM, kn, g);
        }
#pragma unroll
        for (int tt = 0; tt < 4; ++tt)
            acc[tt] = __builtin_amdgcn_wmma_f32_16x16x32_bf16(
                false, a_cur, false, b_cur[tt], (short)0, acc[tt], false, false);
        if (ic < 3) {
            a_cur = a_nxt;
#pragma unroll
            for (int tt = 0; tt < 4; ++tt) b_cur[tt] = b_nxt[tt];
        }
    }

    // Stash per-wave partials (stride-1 in lane -> conflict-free banks).
#pragma unroll
    for (int tt = 0; tt < 4; ++tt)
#pragma unroll
        for (int r = 0; r < 8; ++r)
            red[wave * 1024 + (tt * 8 + r) * 32 + lane] = acc[tt][r];

    __syncthreads();

    // Wave w finalizes tile tt = w&3, row group rg = w>>2 (4 rows each).
    const int tt = wave & 3;
    const int rg = wave >> 2;
    const int n  = nb + tt * 16 + ln;
#pragma unroll
    for (int j = 0; j < 4; ++j) {
        const int r = rg * 4 + j;
        const int base = (tt * 8 + r) * 32 + lane;
        float s = 0.0f;
#pragma unroll
        for (int w = 0; w < 8; ++w) s += red[w * 1024 + base];
        const int m = m0 + r + 8 * g;
        const size_t idx = (size_t)m * L_DIM + n;
        xo[idx] = tanhf(s + xo[idx]);
    }
}

// ---------------------------------------------------------------------------
// Host side
// ---------------------------------------------------------------------------
extern "C" void kernel_launch(void* const* d_in, const int* in_sizes, int n_in,
                              void* d_out, int out_size, void* d_ws, size_t ws_size,
                              hipStream_t stream) {
    (void)in_sizes; (void)n_in; (void)out_size; (void)ws_size;

    const float* x  = (const float*)d_in[0];   // [T, B, I]
    const float* h0 = (const float*)d_in[1];   // [B, L]
    const float* wi = (const float*)d_in[2];   // [L, I]
    const float* bi = (const float*)d_in[3];   // [L]
    const float* wh = (const float*)d_in[4];   // [L, L]
    float* out = (float*)d_out;                // [T, B, L]

    bf16* wib = (bf16*)d_ws;                            // 1 MB
    bf16* whb = wib + (size_t)L_DIM * I_DIM;            // 2 MB

    // One-time (per call) weight conversion to bf16.
    f32_to_bf16_kernel<<<(L_DIM * I_DIM) / 256, 256, 0, stream>>>(wi, wib, L_DIM * I_DIM);
    f32_to_bf16_kernel<<<(L_DIM * L_DIM) / 256, 256, 0, stream>>>(wh, whb, L_DIM * L_DIM);

    // Phase 1: input projection for all timesteps, written into d_out as xi.
    dim3 g1((T_STEPS * BATCH) / 64, L_DIM / 64);   // (512, 16)
    rnn_xproj_kernel<<<g1, 128, 0, stream>>>(x, wib, bi, out);

    // Phase 2: 512 dependent steps, one launch each (graph replay amortizes).
    dim3 g2(BATCH / 16, L_DIM / 64);               // (4, 16)
    for (int t = 0; t < T_STEPS; ++t)
        rnn_step_kernel<<<g2, 256, 0, stream>>>(whb, h0, out, t);
}
